// ConVIRTContrastiveCriterion_42030549958988
// MI455X (gfx1250) — matrix operand order repaired and checked by reading, not compiled, not run on hardware
//
#include <hip/hip_runtime.h>
#include <hip/hip_bf16.h>
#include <math.h>

typedef __attribute__((ext_vector_type(16))) _Float16 v16h;
typedef __attribute__((ext_vector_type(8)))  _Float16 v8h;
typedef __attribute__((ext_vector_type(8)))  float    v8f;

constexpr int N     = 8192;
constexpr int D     = 512;
constexpr int KST   = D / 32;      // 16 k-steps of 32
constexpr int CHUNK = 1024;        // columns per block
constexpr int CT    = CHUNK / 16;  // 64 column tiles per block
constexpr int BPAD  = 8;           // halves of pad per column row (16B)
constexpr int BSTR  = D + BPAD;    // 520 halves: lane i -> banks 4i..4i+3
constexpr float INV_T = 10.0f;     // 1 / TEMPERATURE
constexpr float LAMDA = 0.75f;

// ---------------------------------------------------------------------------
// Row L2-normalize fp32 -> f16.  One wave (32 lanes) per row, 16 elems/lane.
// ---------------------------------------------------------------------------
__global__ __launch_bounds__(256) void normalize_rows(const float* __restrict__ X,
                                                      _Float16* __restrict__ Xh) {
  const int lane = threadIdx.x & 31;
  const int row  = blockIdx.x * 8 + (threadIdx.x >> 5);
  const float* src = X + (size_t)row * D + lane * 16;
  float buf[16];
  float ss = 0.f;
#pragma unroll
  for (int i = 0; i < 16; ++i) { buf[i] = src[i]; ss += buf[i] * buf[i]; }
#pragma unroll
  for (int off = 1; off < 32; off <<= 1) ss += __shfl_xor(ss, off, 32);
  const float inv = 1.0f / fmaxf(sqrtf(ss), 1e-8f);
  _Float16* dst = Xh + (size_t)row * D + lane * 16;
#pragma unroll
  for (int i = 0; i < 16; ++i) dst[i] = (_Float16)(buf[i] * inv);
}

// ---------------------------------------------------------------------------
// Fused GEMM + row-wise sum(exp(sim/T)).
// Block: 8 waves = 128 rows x CHUNK cols.  B tiles (16 cols x K=512, 16KB,
// contiguous in memory) are double-buffered in LDS via async global->LDS DMA
// and shared by all 8 waves.  A fragments are register-resident per wave.
// ---------------------------------------------------------------------------
__global__ __launch_bounds__(256) void simexp_rows(const _Float16* __restrict__ A,
                                                   const _Float16* __restrict__ Bm,
                                                   float* __restrict__ rowsum) {
  __shared__ __align__(16) _Float16 smem[2][16 * BSTR];

  const int tid  = threadIdx.x;
  const int lane = tid & 31;
  const int wave = tid >> 5;
  const int m0   = (blockIdx.y * 8 + wave) * 16;
  const int n0   = blockIdx.x * CHUNK;
  const int half = lane >> 4;
  const int l15  = lane & 15;

  const void* bbase = (const void*)(Bm + (size_t)n0 * D);

  // Issue one 16KB tile (cols n0+ct*16 .. +15, all K) as 4 async 16B
  // loads per thread: 256 thr * 4 * 16B = 16KB.  ASYNCcnt-tracked DMA.
  auto stage = [&](int ct) {
    const int buf = ct & 1;
#pragma unroll
    for (int r = 0; r < 4; ++r) {
      const int c      = r * 256 + tid;        // 16B chunk id in tile
      const int row    = c >> 6;               // source row (column of sim)
      const int within = c & 63;               // 16B chunk within 1KB row
      unsigned lds_off = (unsigned)(size_t)&smem[buf][row * BSTR + within * 8];
      unsigned goff    = (unsigned)(ct * (16 * D * 2) + c * 16);
      asm volatile("global_load_async_to_lds_b128 %0, %1, %2"
                   :: "v"(lds_off), "v"(goff), "s"(bbase) : "memory");
    }
  };

  // A fragment layout (16-bit A 16x32): lanes 0-15 -> K 0..7 & 16..23,
  // lanes 16-31 -> K 8..15 & 24..31, row M = lane&15.
  v16h a[KST];
  const _Float16* arow = A + (size_t)(m0 + l15) * D;
#pragma unroll
  for (int ks = 0; ks < KST; ++ks) {
    const _Float16* p = arow + ks * 32 + half * 8;
    v8h lo = *(const v8h*)(p);
    v8h hi = *(const v8h*)(p + 16);
    a[ks] = __builtin_shufflevector(lo, hi, 0,1,2,3,4,5,6,7,8,9,10,11,12,13,14,15);
  }

  float racc[8];
#pragma unroll
  for (int r = 0; r < 8; ++r) racc[r] = 0.f;

  stage(0);
  for (int ct = 0; ct < CT; ++ct) {
    if (ct + 1 < CT) {
      stage(ct + 1);                                   // prefetch next tile
      asm volatile("s_wait_asynccnt 0x4" ::: "memory"); // tile ct landed
    } else {
      asm volatile("s_wait_asynccnt 0x0" ::: "memory");
    }
    __syncthreads();

    // B fragment (32x16): lane = column, lanes 0-15 -> K 0..7 & 16..23, 16-31 -> K 8..15 & 24..31.
    const _Float16* bcol = &smem[ct & 1][l15 * BSTR + half * 8];
    v8f c = {};
#pragma unroll
    for (int ks = 0; ks < KST; ++ks) {
      v8h lo = *(const v8h*)(bcol + ks * 32);
      v8h hi = *(const v8h*)(bcol + ks * 32 + 16);
      v16h b = __builtin_shufflevector(lo, hi, 0,1,2,3,4,5,6,7,8,9,10,11,12,13,14,15);
      c = __builtin_amdgcn_wmma_f32_16x16x32_f16(false, a[ks], false, b,
                                                 (short)0, c, false, false);
    }
    // C layout: lane l holds (M = r + 8*(l>>4), N = l&15) in c[r].
#pragma unroll
    for (int r = 0; r < 8; ++r) racc[r] += __expf(c[r] * INV_T);

    __syncthreads();   // all waves done reading buf[ct&1] before its reuse
  }

  // Sum over the 16 lanes (= 16 columns) within each half-wave.
#pragma unroll
  for (int r = 0; r < 8; ++r) {
#pragma unroll
    for (int off = 1; off < 16; off <<= 1)
      racc[r] += __shfl_xor(racc[r], off, 32);
  }
  if (l15 == 0) {
#pragma unroll
    for (int r = 0; r < 8; ++r)
      atomicAdd(&rowsum[m0 + half * 8 + r], racc[r]);
  }
}

// ---------------------------------------------------------------------------
// diag_i = dot(v_i, u_i) / T   (from the same f16 operands as the GEMM)
// ---------------------------------------------------------------------------
__global__ __launch_bounds__(256) void diag_kernel(const _Float16* __restrict__ vh,
                                                   const _Float16* __restrict__ uh,
                                                   float* __restrict__ dg) {
  const int lane = threadIdx.x & 31;
  const int row  = blockIdx.x * 8 + (threadIdx.x >> 5);
  const _Float16* pv = vh + (size_t)row * D + lane * 16;
  const _Float16* pu = uh + (size_t)row * D + lane * 16;
  float s = 0.f;
#pragma unroll
  for (int i = 0; i < 16; ++i) s += (float)pv[i] * (float)pu[i];
#pragma unroll
  for (int off = 1; off < 32; off <<= 1) s += __shfl_xor(s, off, 32);
  if (lane == 0) dg[row] = s * INV_T;
}

// ---------------------------------------------------------------------------
// loss = mean_i [ lam*log(rs_v_i) + (1-lam)*log(rs_u_i) - diag_i ]
// ---------------------------------------------------------------------------
__global__ __launch_bounds__(256) void loss_kernel(const float* __restrict__ rs_v,
                                                   const float* __restrict__ rs_u,
                                                   const float* __restrict__ dg,
                                                   float* __restrict__ out) {
  __shared__ float sdata[256];
  float acc = 0.f;
  for (int i = threadIdx.x; i < N; i += 256)
    acc += LAMDA * logf(rs_v[i]) + (1.0f - LAMDA) * logf(rs_u[i]) - dg[i];
  sdata[threadIdx.x] = acc;
  __syncthreads();
  for (int s = 128; s > 0; s >>= 1) {
    if ((int)threadIdx.x < s) sdata[threadIdx.x] += sdata[threadIdx.x + s];
    __syncthreads();
  }
  if (threadIdx.x == 0) out[0] = sdata[0] / (float)N;
}

// ---------------------------------------------------------------------------
extern "C" void kernel_launch(void* const* d_in, const int* in_sizes, int n_in,
                              void* d_out, int out_size, void* d_ws, size_t ws_size,
                              hipStream_t stream) {
  const float* v = (const float*)d_in[0];
  const float* u = (const float*)d_in[1];
  float* out = (float*)d_out;

  char* ws = (char*)d_ws;
  _Float16* vh = (_Float16*)ws;
  _Float16* uh = vh + (size_t)N * D;
  float* rs_v  = (float*)(uh + (size_t)N * D);
  float* rs_u  = rs_v + N;
  float* dg    = rs_u + N;

  // rs_v and rs_u are contiguous: zero both accumulators each call.
  hipMemsetAsync(rs_v, 0, 2 * N * sizeof(float), stream);

  normalize_rows<<<N / 8, 256, 0, stream>>>(v, vh);
  normalize_rows<<<N / 8, 256, 0, stream>>>(u, uh);
  diag_kernel<<<N / 8, 256, 0, stream>>>(vh, uh, dg);

  dim3 grid(N / CHUNK, N / 16 / 8);  // (8, 64), 8 waves/block
  simexp_rows<<<grid, 256, 0, stream>>>(vh, uh, rs_v);  // rows of sim
  simexp_rows<<<grid, 256, 0, stream>>>(uh, vh, rs_u);  // rows of sim^T = cols

  loss_kernel<<<1, 256, 0, stream>>>(rs_v, rs_u, dg, out);
}
